// LoRACrossAttention_48000554500833
// MI455X (gfx1250) — compile-verified, matrix operand order checked by reading, not run on hardware
//
#include <hip/hip_runtime.h>
#include <hip/hip_bf16.h>

#define DIM      768
#define HEADS    12
#define HEAD_DIM 64
#define RANK     8
#define BATCH    8
#define SEQ      1024
#define ROWS     (BATCH * SEQ)          /* 8192 */
#define SM_SCALE 0.125f                 /* 64^-0.5 */
#define LORA_SC  0.125f                 /* 1/RANK  */
#define LDSW     1032                   /* padded LDS row stride (floats); 1032%64=8 */

typedef __attribute__((ext_vector_type(16))) _Float16 v16h;
typedef __attribute__((ext_vector_type(8)))  _Float16 v8h;
typedef __attribute__((ext_vector_type(8)))  float    v8f;
typedef __attribute__((ext_vector_type(4)))  float    v4f;

// ---------------------------------------------------------------------------
// Fragment helpers (CDNA5 wave32 WMMA 16x16x32 f16 layouts, ISA 7.12.2)
//   A/B lane L: index = L&15, half-sel = L>>4
//   h[i<8]  = K = 8*hsel + i ;  h[i>=8] = K = 16 + 8*hsel + (i-8)
//   C/D lane L: col = L&15, row = v + 8*(L>>4)
// ---------------------------------------------------------------------------
__device__ __forceinline__ v16h frag_from_f16(const _Float16* p) {
    v8h lo = *(const v8h*)(p);
    v8h hi = *(const v8h*)(p + 16);
    v16h f;
#pragma unroll
    for (int i = 0; i < 8; ++i) { f[i] = lo[i]; f[8 + i] = hi[i]; }
    return f;
}

__device__ __forceinline__ v16h frag_from_f32(const float* p) {
    v16h f;
#pragma unroll
    for (int i = 0; i < 8; ++i) {
        f[i]     = (_Float16)p[i];
        f[8 + i] = (_Float16)p[16 + i];
    }
    return f;
}

// ---------------------------------------------------------------------------
// Kernel 1: fold LoRA into weights, convert to f16
// ---------------------------------------------------------------------------
__global__ void __launch_bounds__(256)
prep_weights(const float* __restrict__ Wq, const float* __restrict__ Aq, const float* __restrict__ Bq,
             const float* __restrict__ Wk,
             const float* __restrict__ Wv, const float* __restrict__ Av, const float* __restrict__ Bv,
             const float* __restrict__ Wp,
             _Float16* __restrict__ wq, _Float16* __restrict__ wk,
             _Float16* __restrict__ wv, _Float16* __restrict__ wp) {
    int idx = blockIdx.x * 256 + threadIdx.x;
    if (idx >= DIM * DIM) return;
    int n = idx / DIM, k = idx % DIM;
    int z = blockIdx.z;
    if (z == 0) {
        float v = Wq[idx];
#pragma unroll
        for (int r = 0; r < RANK; ++r) v += Bq[n * RANK + r] * Aq[r * DIM + k] * LORA_SC;
        wq[idx] = (_Float16)v;
    } else if (z == 1) {
        wk[idx] = (_Float16)Wk[idx];
    } else if (z == 2) {
        float v = Wv[idx];
#pragma unroll
        for (int r = 0; r < RANK; ++r) v += Bv[n * RANK + r] * Av[r * DIM + k] * LORA_SC;
        wv[idx] = (_Float16)v;
    } else {
        wp[idx] = (_Float16)Wp[idx];
    }
}

// ---------------------------------------------------------------------------
// Kernel 2: projections  Y = X @ W^T   (X f32 [8192,768], W f16 [768,768])
//   z=0: qh -> [b,h,n,64] ; z=1: kh -> [b,h,n,64] ; z=2: vT -> [b,h,64,n]
// One wave per 16(M) x 64(N) tile; K = 768 in steps of 32.
// ---------------------------------------------------------------------------
__global__ void __launch_bounds__(32)
gemm_proj(const float* __restrict__ q, const float* __restrict__ x,
          const _Float16* __restrict__ wq, const _Float16* __restrict__ wk,
          const _Float16* __restrict__ wv,
          _Float16* __restrict__ qh, _Float16* __restrict__ kh, _Float16* __restrict__ vT) {
    const int lane = threadIdx.x & 31;
    const int ln   = lane & 15;
    const int hsel = lane >> 4;
    const int m0   = blockIdx.x * 16;
    const int n0   = blockIdx.y * 64;
    const int z    = blockIdx.z;

    const float*    X = (z == 0) ? q : x;
    const _Float16* W = (z == 0) ? wq : (z == 1) ? wk : wv;

    v8f acc[4] = {};
    const float* xrow = X + (size_t)(m0 + ln) * DIM;
#pragma unroll 2
    for (int kk = 0; kk < DIM; kk += 32) {
        const int kb = kk + 8 * hsel;
        v16h a = frag_from_f32(xrow + kb);
        v16h bf[4];
#pragma unroll
        for (int j = 0; j < 4; ++j)
            bf[j] = frag_from_f16(W + (size_t)(n0 + j * 16 + ln) * DIM + kb);
#pragma unroll
        for (int j = 0; j < 4; ++j)
            acc[j] = __builtin_amdgcn_wmma_f32_16x16x32_f16(false, a, false, bf[j],
                                                            (short)0, acc[j], false, false);
    }

#pragma unroll
    for (int j = 0; j < 4; ++j) {
#pragma unroll
        for (int v = 0; v < 8; ++v) {
            const int f   = n0 + j * 16 + ln;            // output feature 0..767
            const int hh  = f >> 6, dd = f & 63;
            const int row = m0 + v + 8 * hsel;           // global seq row 0..8191
            const int b   = row >> 10, ns = row & 1023;
            const _Float16 val = (_Float16)acc[j][v];
            if (z == 2)
                vT[((size_t)(b * HEADS + hh) * HEAD_DIM + dd) * SEQ + ns] = val;
            else if (z == 0)
                qh[((size_t)(b * HEADS + hh) * SEQ + ns) * HEAD_DIM + dd] = val;
            else
                kh[((size_t)(b * HEADS + hh) * SEQ + ns) * HEAD_DIM + dd] = val;
        }
    }
}

// ---------------------------------------------------------------------------
// Kernel 3 (FUSED): per (b,h, 16-row strip):
//   S = q@k^T in LDS -> softmax stats -> normalized attn written once to
//   global -> ctx = P@v straight from LDS (normalization folded into epilogue)
// 128 threads = 4 waves. LDS strip: 16 x 1032 f32 (~66 KB of the 320 KB WGP).
// ---------------------------------------------------------------------------
__global__ void __launch_bounds__(128)
fused_attn(const _Float16* __restrict__ qh, const _Float16* __restrict__ kh,
           const _Float16* __restrict__ vT,
           float* __restrict__ attn, _Float16* __restrict__ ctx) {
    __shared__ float S[16 * LDSW];
    __shared__ float red[16][8];
    __shared__ float rowmax[16];
    __shared__ float rowinv[16];

    const int tid  = threadIdx.x;
    const int lane = tid & 31;
    const int wv   = tid >> 5;          // wave 0..3
    const int ln   = lane & 15;
    const int hsel = lane >> 4;
    const int m0   = blockIdx.x * 16;   // n1 strip
    const int bh   = blockIdx.y;        // 0..95
    const int b    = bh / HEADS, h = bh % HEADS;

    const _Float16* qb = qh + (size_t)bh * SEQ * HEAD_DIM;
    const _Float16* kb = kh + (size_t)bh * SEQ * HEAD_DIM;
    const _Float16* vb = vT + (size_t)bh * HEAD_DIM * SEQ;

    // ---- Phase 1: raw scores into LDS. Wave wv covers n2 in [wv*256, wv*256+256)
    const v16h a0 = frag_from_f16(qb + (size_t)(m0 + ln) * HEAD_DIM + 8 * hsel);
    const v16h a1 = frag_from_f16(qb + (size_t)(m0 + ln) * HEAD_DIM + 32 + 8 * hsel);
#pragma unroll
    for (int s = 0; s < 4; ++s) {
        const int n0 = wv * 256 + s * 64;
        v8f acc[4] = {};
        v16h b0[4], b1[4];
#pragma unroll
        for (int j = 0; j < 4; ++j) {
            const _Float16* kr = kb + (size_t)(n0 + j * 16 + ln) * HEAD_DIM;
            b0[j] = frag_from_f16(kr + 8 * hsel);
            b1[j] = frag_from_f16(kr + 32 + 8 * hsel);
        }
#pragma unroll
        for (int j = 0; j < 4; ++j) {
            acc[j] = __builtin_amdgcn_wmma_f32_16x16x32_f16(false, a0, false, b0[j],
                                                            (short)0, acc[j], false, false);
            acc[j] = __builtin_amdgcn_wmma_f32_16x16x32_f16(false, a1, false, b1[j],
                                                            (short)0, acc[j], false, false);
        }
#pragma unroll
        for (int j = 0; j < 4; ++j)
#pragma unroll
            for (int v = 0; v < 8; ++v)
                S[(v + 8 * hsel) * LDSW + n0 + j * 16 + ln] = acc[j][v];
    }
    __syncthreads();

    // ---- Phase 2: softmax stats. 8 threads per row, 128 cols each.
    const int r   = tid >> 3;
    const int seg = tid & 7;
    float* srow = S + r * LDSW + seg * 128;

    float mx = -3.0e38f;
#pragma unroll 8
    for (int i = 0; i < 128; ++i) mx = fmaxf(mx, srow[i]);
    red[r][seg] = mx;
    __syncthreads();
    if (seg == 0) {
        float m = red[r][0];
#pragma unroll
        for (int i = 1; i < 8; ++i) m = fmaxf(m, red[r][i]);
        rowmax[r] = m;
    }
    __syncthreads();

    const float m = rowmax[r];
    float sum = 0.0f;
#pragma unroll 8
    for (int i = 0; i < 128; ++i) {
        float e = __expf((srow[i] - m) * SM_SCALE);
        srow[i] = e;                    // keep unnormalized exp in LDS
        sum += e;
    }
    red[r][seg] = sum;
    __syncthreads();
    if (seg == 0) {
        float t = 0.0f;
#pragma unroll
        for (int i = 0; i < 8; ++i) t += red[r][i];
        rowinv[r] = 1.0f / t;
    }
    __syncthreads();

    // ---- Phase 3: write normalized attn to global (only materialization)
    {
        const float inv_r = rowinv[r];
        float* grow = attn + (size_t)bh * SEQ * SEQ + (size_t)(m0 + r) * SEQ + seg * 128;
#pragma unroll 8
        for (int i = 0; i < 128; i += 4) {
            v4f e = *(const v4f*)(srow + i);
            e = e * inv_r;
            *(v4f*)(grow + i) = e;
        }
    }

    // ---- Phase 4: ctx tile = (Pexp @ v) * rowinv. Wave wv does d-cols [wv*16, wv*16+16)
    v8f cacc = {};
    const float* pr = S + ln * LDSW;
#pragma unroll 2
    for (int kk = 0; kk < SEQ; kk += 32) {
        const int kbs = kk + 8 * hsel;
        v16h a = frag_from_f32(pr + kbs);
        v16h bf = frag_from_f16(vb + (size_t)(wv * 16 + ln) * SEQ + kbs);
        cacc = __builtin_amdgcn_wmma_f32_16x16x32_f16(false, a, false, bf,
                                                      (short)0, cacc, false, false);
    }
#pragma unroll
    for (int v = 0; v < 8; ++v) {
        const int row = v + 8 * hsel;
        const float val = cacc[v] * rowinv[row];
        ctx[(size_t)(b * SEQ + m0 + row) * DIM + h * HEAD_DIM + wv * 16 + ln] =
            (_Float16)val;
    }
}

// ---------------------------------------------------------------------------
// Kernel 4: out = ctx @ Wp^T + bp   (f16 x f16 -> f32), [8192,768]
// ---------------------------------------------------------------------------
__global__ void __launch_bounds__(32)
out_proj(const _Float16* __restrict__ ctx, const _Float16* __restrict__ wp,
         const float* __restrict__ bias, float* __restrict__ out) {
    const int lane = threadIdx.x & 31;
    const int ln   = lane & 15;
    const int hsel = lane >> 4;
    const int m0   = blockIdx.x * 16;
    const int n0   = blockIdx.y * 64;

    const _Float16* arow = ctx + (size_t)(m0 + ln) * DIM;
    v8f acc[4] = {};
#pragma unroll 2
    for (int kk = 0; kk < DIM; kk += 32) {
        const int kb = kk + 8 * hsel;
        v16h a = frag_from_f16(arow + kb);
        v16h bf[4];
#pragma unroll
        for (int j = 0; j < 4; ++j)
            bf[j] = frag_from_f16(wp + (size_t)(n0 + j * 16 + ln) * DIM + kb);
#pragma unroll
        for (int j = 0; j < 4; ++j)
            acc[j] = __builtin_amdgcn_wmma_f32_16x16x32_f16(false, a, false, bf[j],
                                                            (short)0, acc[j], false, false);
    }

#pragma unroll
    for (int j = 0; j < 4; ++j)
#pragma unroll
        for (int v = 0; v < 8; ++v) {
            const int c = n0 + j * 16 + ln;
            const int rr = m0 + v + 8 * hsel;
            out[(size_t)rr * DIM + c] = acc[j][v] + bias[c];
        }
}

// ---------------------------------------------------------------------------
extern "C" void kernel_launch(void* const* d_in, const int* in_sizes, int n_in,
                              void* d_out, int out_size, void* d_ws, size_t ws_size,
                              hipStream_t stream) {
    const float* q  = (const float*)d_in[0];
    const float* x  = (const float*)d_in[1];
    const float* Wq = (const float*)d_in[2];
    const float* Aq = (const float*)d_in[3];
    const float* Bq = (const float*)d_in[4];
    const float* Wk = (const float*)d_in[5];
    const float* Wv = (const float*)d_in[6];
    const float* Av = (const float*)d_in[7];
    const float* Bv = (const float*)d_in[8];
    const float* Wp = (const float*)d_in[9];
    const float* bp = (const float*)d_in[10];

    float* out  = (float*)d_out;                 // [8,1024,768]
    float* attn = out + (size_t)ROWS * DIM;      // [8,12,1024,1024]

    // workspace layout (f16 elements), ~55 MB total
    _Float16* ws = (_Float16*)d_ws;
    const size_t WSZ = (size_t)DIM * DIM;        // 589824
    const size_t HSZ = (size_t)ROWS * DIM;       // 6291456
    _Float16* wq  = ws;
    _Float16* wk  = wq + WSZ;
    _Float16* wv  = wk + WSZ;
    _Float16* wp  = wv + WSZ;
    _Float16* qh  = wp + WSZ;
    _Float16* kh  = qh + HSZ;
    _Float16* vT  = kh + HSZ;
    _Float16* ctx = vT + HSZ;

    prep_weights<<<dim3((DIM * DIM + 255) / 256, 1, 4), 256, 0, stream>>>(
        Wq, Aq, Bq, Wk, Wv, Av, Bv, Wp, wq, wk, wv, wp);

    gemm_proj<<<dim3(ROWS / 16, DIM / 64, 3), 32, 0, stream>>>(
        q, x, wq, wk, wv, qh, kh, vT);

    fused_attn<<<dim3(SEQ / 16, BATCH * HEADS), 128, 0, stream>>>(
        qh, kh, vT, attn, ctx);

    out_proj<<<dim3(ROWS / 16, DIM / 64), 32, 0, stream>>>(ctx, wp, bp, out);
}
